// GAT_3p_81063212744714
// MI455X (gfx1250) — matrix operand order repaired and checked by reading, not compiled, not run on hardware
//
#include <hip/hip_runtime.h>
#include <hip/hip_bf16.h>

// ---------------- problem constants (match reference) ----------------
#define NN   50000
#define EE   800000
#define GG   64
#define IN_F 128
#define H1_  4
#define C1_  32
#define D1_  128   // H1*C1
#define C2_  64
#define OUTF 16
#define EPSV 1e-5f
#define KK_  128   // K for both layer GEMMs

typedef __attribute__((ext_vector_type(16))) __bf16         v16bf;
typedef __attribute__((ext_vector_type(8)))  float          v8f;
typedef __attribute__((ext_vector_type(8)))  unsigned short v8us;

// ---------------- helpers ----------------
__device__ __forceinline__ unsigned short bf16bits(float f) {
    union { float f; unsigned u; } v; v.f = f;
    unsigned r = v.u + 0x7FFFu + ((v.u >> 16) & 1u);   // round-to-nearest-even
    return (unsigned short)(r >> 16);
}

// order-preserving float <-> uint for atomicMax-based segment_max
__device__ __forceinline__ unsigned fenc(float f) {
    unsigned u = __float_as_uint(f);
    return (u & 0x80000000u) ? ~u : (u | 0x80000000u);
}
__device__ __forceinline__ float fdec(unsigned u) {
    return __uint_as_float((u & 0x80000000u) ? (u & 0x7fffffffu) : ~u);
}
#define NEGINF_ENC 0x007FFFFFu   // fenc(-inf)

// ---------------- generic fills / converts ----------------
__global__ void fill_f32_kernel(float* p, float v, int n) {
    int i = blockIdx.x * blockDim.x + threadIdx.x;
    if (i < n) p[i] = v;
}
__global__ void fill_u32_kernel(unsigned* p, unsigned v, int n) {
    int i = blockIdx.x * blockDim.x + threadIdx.x;
    if (i < n) p[i] = v;
}
// row-major f32 -> bf16 bits
__global__ void cvt_bf16_kernel(const float* __restrict__ in, unsigned short* __restrict__ out, int n) {
    int i = blockIdx.x * blockDim.x + threadIdx.x;
    if (i < n) out[i] = bf16bits(in[i]);
}
// W[K,NC] (f32, row-major) -> Wt[NC,K] (bf16, row-major) : transpose + convert
__global__ void wt_cvt_kernel(const float* __restrict__ W, unsigned short* __restrict__ Wt,
                              int K, int NC) {
    int i = blockIdx.x * blockDim.x + threadIdx.x;
    if (i >= K * NC) return;
    int k = i / NC, n = i - k * NC;
    Wt[(size_t)n * K + k] = bf16bits(W[i]);
}

// ---------------- WMMA GEMM: C[M,NC] = A[M,128] @ W[128,NC] + bias ----------------
// A pre-converted to bf16 row-major [M,128]; W pre-transposed+converted to bf16 [NC,128].
// One 16x16 output tile per wave; K=128 fully unrolled -> 4 v_wmma per tile.
__global__ void wmma_gemm_bias_bf16_kernel(const unsigned short* __restrict__ Abf,
                                           const unsigned short* __restrict__ WtBf,
                                           const float* __restrict__ bias,
                                           float* __restrict__ C,
                                           int NC, int tilesTotal) {
    int waveId = blockIdx.x * (blockDim.x >> 5) + (threadIdx.x >> 5);
    if (waveId >= tilesTotal) return;          // wave-uniform exit, EXEC stays all-1s
    int lane    = threadIdx.x & 31;
    int halfsel = lane >> 4;                   // 0: lanes 0-15, 1: lanes 16-31
    int lid     = lane & 15;
    int tilesN  = NC >> 4;
    int tm = waveId / tilesN;
    int tn = waveId - tm * tilesN;

    int rowA = tm * 16 + lid;                  // A fragment: lane = row
    int colB = tn * 16 + lid;                  // B fragment: lane = column
    // group granularity: 8 bf16 = 16 bytes
    const v8us* Ap = (const v8us*)(Abf  + (size_t)rowA * KK_);
    const v8us* Wp = (const v8us*)(WtBf + (size_t)colB * KK_);

    v8f acc = {};
    #pragma unroll
    for (int k0 = 0; k0 < KK_; k0 += 32) {
        int g = k0 >> 3;
        union { v16bf v; v8us h[2]; } a, b;
        // A 16x32 bf16: half 0 -> K{k0..k0+7, k0+16..k0+23}; half 1 -> K{k0+8.., k0+24..}
        a.h[0] = Ap[g + halfsel];
        a.h[1] = Ap[g + 2 + halfsel];
        // B 32x16 bf16 (from transposed W): half 0 -> K{k0..k0+7, k0+16..k0+23}; half 1 -> K{k0+8.., k0+24..}
        b.h[0] = Wp[g + halfsel];
        b.h[1] = Wp[g + 2 + halfsel];
        acc = __builtin_amdgcn_wmma_f32_16x16x32_bf16(
                  false, a.v, false, b.v, (short)0, acc, false, false);
    }
    float bv = bias[colB];
    #pragma unroll
    for (int v = 0; v < 8; ++v) {              // C/D: VGPR v -> row v + 8*halfsel
        int row = tm * 16 + v + 8 * halfsel;
        C[(size_t)row * NC + colB] = acc[v] + bv;
    }
}

// ---------------- attention logits: alpha[e,h] = sum_c leaky(xl[s]+xr[d]+ea*We)*att ----------------
__global__ void alpha_kernel(const float* __restrict__ xl, const float* __restrict__ xr,
                             const float* __restrict__ edge_attr, const float* __restrict__ We,
                             const float* __restrict__ att,
                             const int* __restrict__ src, const int* __restrict__ dst,
                             float* __restrict__ alpha, int nE, int H, int Cc) {
    int idx = blockIdx.x * blockDim.x + threadIdx.x;
    if (idx >= nE * H) return;
    int e = idx / H, h = idx - e * H;
    int s = src[e], d = dst[e];
    float ea = edge_attr[e];
    int D = H * Cc;
    const float* xls = xl + (size_t)s * D + h * Cc;
    const float* xrd = xr + (size_t)d * D + h * Cc;
    const float* Wh  = We + h * Cc;
    const float* ah  = att + h * Cc;
    float sum = 0.f;
    for (int c = 0; c < Cc; ++c) {
        float m = xls[c] + xrd[c] + ea * Wh[c];
        m = (m > 0.f) ? m : 0.2f * m;          // leaky_relu(0.2)
        sum += m * ah[c];
    }
    alpha[(size_t)e * H + h] = sum;
}

// ---------------- segment max of alpha over dst ----------------
__global__ void amax_kernel(const float* __restrict__ alpha, const int* __restrict__ dst,
                            unsigned* __restrict__ amax, int nE, int H) {
    int idx = blockIdx.x * blockDim.x + threadIdx.x;
    if (idx >= nE * H) return;
    int e = idx / H, h = idx - e * H;
    atomicMax(&amax[(size_t)dst[e] * H + h], fenc(alpha[idx]));
}

// ---------------- softmax denominator ----------------
__global__ void denom_kernel(const float* __restrict__ alpha, const int* __restrict__ dst,
                             const unsigned* __restrict__ amax, float* __restrict__ denom,
                             int nE, int H) {
    int idx = blockIdx.x * blockDim.x + threadIdx.x;
    if (idx >= nE * H) return;
    int e = idx / H, h = idx - e * H;
    unsigned u = amax[(size_t)dst[e] * H + h];
    float am = (u == NEGINF_ENC) ? 0.f : fdec(u);
    atomicAdd(&denom[(size_t)dst[e] * H + h], __expf(alpha[idx] - am));
}

// ---------------- weighted aggregation: out[dst] += xl[src] * a ----------------
__global__ void aggregate_kernel(const float* __restrict__ xl, const float* __restrict__ alpha,
                                 const unsigned* __restrict__ amax, const float* __restrict__ denom,
                                 const int* __restrict__ src, const int* __restrict__ dst,
                                 float* __restrict__ out, int nE, int H, int Cc) {
    int D = H * Cc;
    int idx = blockIdx.x * blockDim.x + threadIdx.x;
    if (idx >= nE * D) return;
    int e = idx / D, c = idx - e * D;
    int h = c / Cc;
    int d = dst[e];
    unsigned u = amax[(size_t)d * H + h];
    float am = (u == NEGINF_ENC) ? 0.f : fdec(u);
    float p  = __expf(alpha[(size_t)e * H + h] - am);
    float a  = p / (denom[(size_t)d * H + h] + 1e-16f);
    atomicAdd(&out[(size_t)d * D + c], xl[(size_t)src[e] * D + c] * a);
}

// ---------------- h = relu(h + bias) ----------------
__global__ void bias_relu_kernel(float* __restrict__ h, const float* __restrict__ bias,
                                 int n, int D) {
    int idx = blockIdx.x * blockDim.x + threadIdx.x;
    if (idx >= n * D) return;
    float v = h[idx] + bias[idx % D];
    h[idx] = (v > 0.f) ? v : 0.f;
}

// ---------------- batchnorm stats (one block per channel) ----------------
__global__ void bn_stats_kernel(const float* __restrict__ h, float* __restrict__ mean,
                                float* __restrict__ var, int n, int D) {
    int c = blockIdx.x, t = threadIdx.x;
    float s = 0.f, s2 = 0.f;
    for (int r = t; r < n; r += blockDim.x) {
        float v = h[(size_t)r * D + c];
        s += v; s2 += v * v;
    }
    __shared__ float sh[256], sh2[256];
    sh[t] = s; sh2[t] = s2; __syncthreads();
    for (int off = 128; off > 0; off >>= 1) {
        if (t < off) { sh[t] += sh[t + off]; sh2[t] += sh2[t + off]; }
        __syncthreads();
    }
    if (t == 0) {
        float mu = sh[0] / (float)n;
        mean[c] = mu;
        var[c]  = sh2[0] / (float)n - mu * mu;   // biased variance
    }
}

__global__ void bn_apply_kernel(float* __restrict__ h, const float* __restrict__ mean,
                                const float* __restrict__ var, const float* __restrict__ gamma,
                                const float* __restrict__ beta, int n, int D) {
    int idx = blockIdx.x * blockDim.x + threadIdx.x;
    if (idx >= n * D) return;
    int c = idx % D;
    h[idx] = (h[idx] - mean[c]) * rsqrtf(var[c] + EPSV) * gamma[c] + beta[c];
}

// ---------------- graph pooling: sum / count / max ----------------
__global__ void pool_kernel(const float* __restrict__ h, const int* __restrict__ batch,
                            float* __restrict__ s, float* __restrict__ cnt,
                            unsigned* __restrict__ mx, int n, int D) {
    int idx = blockIdx.x * blockDim.x + threadIdx.x;
    if (idx >= n * D) return;
    int r = idx / D, c = idx - r * D;
    int b = batch[r];
    float v = h[idx];
    atomicAdd(&s[(size_t)b * D + c], v);
    atomicMax(&mx[(size_t)b * D + c], fenc(v));
    if (c == 0) atomicAdd(&cnt[b], 1.f);
}

__global__ void feat_kernel(const float* __restrict__ s, const float* __restrict__ cnt,
                            const unsigned* __restrict__ mx, float* __restrict__ feat) {
    int idx = blockIdx.x * blockDim.x + threadIdx.x;
    if (idx >= GG * 3 * C2_) return;
    int g = idx / (3 * C2_), j = idx - g * 3 * C2_;
    float v;
    if (j < C2_)            v = s[(size_t)g * C2_ + j];
    else if (j < 2 * C2_)   v = s[(size_t)g * C2_ + (j - C2_)] / fmaxf(cnt[g], 1.f);
    else                    v = fdec(mx[(size_t)g * C2_ + (j - 2 * C2_)]); // -inf for empty, as segment_max
    feat[idx] = v;
}

// ---------------- final linear: out[G,16] = feat[G,192] @ Wlin + blin ----------------
__global__ void linear_kernel(const float* __restrict__ feat, const float* __restrict__ Wlin,
                              const float* __restrict__ blin, float* __restrict__ out) {
    int idx = blockIdx.x * blockDim.x + threadIdx.x;
    if (idx >= GG * OUTF) return;
    int g = idx / OUTF, o = idx - g * OUTF;
    float sum = blin[o];
    const float* fg = feat + (size_t)g * 3 * C2_;
    for (int j = 0; j < 3 * C2_; ++j) sum += fg[j] * Wlin[(size_t)j * OUTF + o];
    out[idx] = sum;
}

// ---------------- launch ----------------
static inline int cdiv(long long a, int b) { return (int)((a + b - 1) / b); }

extern "C" void kernel_launch(void* const* d_in, const int* in_sizes, int n_in,
                              void* d_out, int out_size, void* d_ws, size_t ws_size,
                              hipStream_t stream) {
    (void)in_sizes; (void)n_in; (void)out_size; (void)ws_size;

    const float* x         = (const float*)d_in[0];
    const int*   ei        = (const int*)  d_in[1];
    const int*   src       = ei;
    const int*   dst       = ei + EE;
    const float* edge_attr = (const float*)d_in[2];
    const int*   batch     = (const int*)  d_in[3];
    const float* Wl1  = (const float*)d_in[4];  const float* bl1   = (const float*)d_in[5];
    const float* Wr1  = (const float*)d_in[6];  const float* br1   = (const float*)d_in[7];
    const float* We1  = (const float*)d_in[8];  const float* att1  = (const float*)d_in[9];
    const float* bias1= (const float*)d_in[10];
    const float* Wl2  = (const float*)d_in[11]; const float* bl2   = (const float*)d_in[12];
    const float* Wr2  = (const float*)d_in[13]; const float* br2   = (const float*)d_in[14];
    const float* We2  = (const float*)d_in[15]; const float* att2  = (const float*)d_in[16];
    const float* bias2= (const float*)d_in[17];
    const float* bn1g = (const float*)d_in[18]; const float* bn1b  = (const float*)d_in[19];
    const float* bn2g = (const float*)d_in[20]; const float* bn2b  = (const float*)d_in[21];
    const float* Wlin = (const float*)d_in[22]; const float* blin  = (const float*)d_in[23];
    float* out = (float*)d_out;

    // ---- workspace layout (16B alignment preserved: every chunk is a multiple of 4 floats) ----
    float*    ws    = (float*)d_ws;
    float*    xl1   = ws;                              // N*128
    float*    xr1   = xl1 + (size_t)NN * D1_;          // N*128
    float*    h1    = xr1 + (size_t)NN * D1_;          // N*128 (reused as h2: N*64)
    float*    alpha = h1  + (size_t)NN * D1_;          // E*4  (reused as E*1)
    unsigned* amax  = (unsigned*)(alpha + (size_t)EE * H1_); // N*4
    float*    denom = (float*)(amax + (size_t)NN * H1_);     // N*4
    float*    mean  = denom + (size_t)NN * H1_;        // 128
    float*    var   = mean + D1_;                      // 128
    float*    psum  = var + D1_;                       // G*64
    float*    pcnt  = psum + (size_t)GG * C2_;         // G
    unsigned* pmx   = (unsigned*)(pcnt + GG);          // G*64
    float*    feat  = (float*)(pmx + (size_t)GG * C2_);// G*192
    unsigned short* abf = (unsigned short*)(feat + (size_t)GG * 3 * C2_); // N*128 bf16
    unsigned short* wtA = abf + (size_t)NN * KK_;      // 128*128 bf16 (transposed Wl)
    unsigned short* wtB = wtA + (size_t)KK_ * D1_;     // 128*128 bf16 (transposed Wr)
    float*    h2    = h1;                              // alias (h1 dead after GEMM2)
    float*    xl2   = xl1;                             // alias, N*64 stride
    float*    xr2   = xr1;

    const int B = 256;

    // ===== Layer 1: GATv2 (H=4, C=32, concat) =====
    cvt_bf16_kernel<<<cdiv((long long)NN * IN_F, B), B, 0, stream>>>(x, abf, NN * IN_F);
    wt_cvt_kernel<<<cdiv((long long)KK_ * D1_, B), B, 0, stream>>>(Wl1, wtA, KK_, D1_);
    wt_cvt_kernel<<<cdiv((long long)KK_ * D1_, B), B, 0, stream>>>(Wr1, wtB, KK_, D1_);
    {
        int tiles = (NN / 16) * (D1_ / 16);
        wmma_gemm_bias_bf16_kernel<<<cdiv(tiles, 8), B, 0, stream>>>(abf, wtA, bl1, xl1, D1_, tiles);
        wmma_gemm_bias_bf16_kernel<<<cdiv(tiles, 8), B, 0, stream>>>(abf, wtB, br1, xr1, D1_, tiles);
    }
    fill_u32_kernel<<<cdiv((long long)NN * H1_, B), B, 0, stream>>>(amax, NEGINF_ENC, NN * H1_);
    fill_f32_kernel<<<cdiv((long long)NN * H1_, B), B, 0, stream>>>(denom, 0.f, NN * H1_);
    fill_f32_kernel<<<cdiv((long long)NN * D1_, B), B, 0, stream>>>(h1, 0.f, NN * D1_);

    alpha_kernel<<<cdiv((long long)EE * H1_, B), B, 0, stream>>>(xl1, xr1, edge_attr, We1, att1,
                                                                 src, dst, alpha, EE, H1_, C1_);
    amax_kernel <<<cdiv((long long)EE * H1_, B), B, 0, stream>>>(alpha, dst, amax, EE, H1_);
    denom_kernel<<<cdiv((long long)EE * H1_, B), B, 0, stream>>>(alpha, dst, amax, denom, EE, H1_);
    aggregate_kernel<<<cdiv((long long)EE * D1_, B), B, 0, stream>>>(xl1, alpha, amax, denom,
                                                                     src, dst, h1, EE, H1_, C1_);
    bias_relu_kernel<<<cdiv((long long)NN * D1_, B), B, 0, stream>>>(h1, bias1, NN, D1_);
    bn_stats_kernel<<<D1_, B, 0, stream>>>(h1, mean, var, NN, D1_);
    bn_apply_kernel<<<cdiv((long long)NN * D1_, B), B, 0, stream>>>(h1, mean, var, bn1g, bn1b, NN, D1_);

    // ===== Layer 2: GATv2 (H=1, C=64, mean==identity) =====
    cvt_bf16_kernel<<<cdiv((long long)NN * D1_, B), B, 0, stream>>>(h1, abf, NN * D1_);
    wt_cvt_kernel<<<cdiv((long long)KK_ * C2_, B), B, 0, stream>>>(Wl2, wtA, KK_, C2_);
    wt_cvt_kernel<<<cdiv((long long)KK_ * C2_, B), B, 0, stream>>>(Wr2, wtB, KK_, C2_);
    {
        int tiles = (NN / 16) * (C2_ / 16);
        wmma_gemm_bias_bf16_kernel<<<cdiv(tiles, 8), B, 0, stream>>>(abf, wtA, bl2, xl2, C2_, tiles);
        wmma_gemm_bias_bf16_kernel<<<cdiv(tiles, 8), B, 0, stream>>>(abf, wtB, br2, xr2, C2_, tiles);
    }
    fill_u32_kernel<<<cdiv((long long)NN, B), B, 0, stream>>>(amax, NEGINF_ENC, NN);
    fill_f32_kernel<<<cdiv((long long)NN, B), B, 0, stream>>>(denom, 0.f, NN);
    fill_f32_kernel<<<cdiv((long long)NN * C2_, B), B, 0, stream>>>(h2, 0.f, NN * C2_);

    alpha_kernel<<<cdiv((long long)EE, B), B, 0, stream>>>(xl2, xr2, edge_attr, We2, att2,
                                                           src, dst, alpha, EE, 1, C2_);
    amax_kernel <<<cdiv((long long)EE, B), B, 0, stream>>>(alpha, dst, amax, EE, 1);
    denom_kernel<<<cdiv((long long)EE, B), B, 0, stream>>>(alpha, dst, amax, denom, EE, 1);
    aggregate_kernel<<<cdiv((long long)EE * C2_, B), B, 0, stream>>>(xl2, alpha, amax, denom,
                                                                     src, dst, h2, EE, 1, C2_);
    bias_relu_kernel<<<cdiv((long long)NN * C2_, B), B, 0, stream>>>(h2, bias2, NN, C2_);
    bn_stats_kernel<<<C2_, B, 0, stream>>>(h2, mean, var, NN, C2_);
    bn_apply_kernel<<<cdiv((long long)NN * C2_, B), B, 0, stream>>>(h2, mean, var, bn2g, bn2b, NN, C2_);

    // ===== Pool + final linear =====
    fill_f32_kernel<<<cdiv((long long)GG * C2_, B), B, 0, stream>>>(psum, 0.f, GG * C2_);
    fill_f32_kernel<<<1, B, 0, stream>>>(pcnt, 0.f, GG);
    fill_u32_kernel<<<cdiv((long long)GG * C2_, B), B, 0, stream>>>(pmx, NEGINF_ENC, GG * C2_);

    pool_kernel<<<cdiv((long long)NN * C2_, B), B, 0, stream>>>(h2, batch, psum, pcnt, pmx, NN, C2_);
    feat_kernel<<<cdiv((long long)GG * 3 * C2_, B), B, 0, stream>>>(psum, pcnt, pmx, feat);
    linear_kernel<<<cdiv((long long)GG * OUTF, B), B, 0, stream>>>(feat, Wlin, blin, out);
}
